// BiLSTMCRFPOSTagger_979252544177
// MI455X (gfx1250) — compile-verified, matrix operand order checked by reading, not run on hardware
//
#include <hip/hip_runtime.h>
#include <hip/hip_fp16.h>

typedef __attribute__((ext_vector_type(16))) _Float16 v16h;
typedef __attribute__((ext_vector_type(8)))  _Float16 v8h;
typedef __attribute__((ext_vector_type(8)))  float    v8f;

#define T_LEN    512
#define BATCH    64
#define EDIM     256
#define HDIM     256
#define KTAG     48
#define START_ID 46
#define NEG_INF  (-10000.0f)

// ---- optional CDNA5 async memory->LDS path (ASYNCcnt-tracked DMA) ----------
#if defined(__has_builtin)
#if __has_builtin(__builtin_amdgcn_global_load_async_to_lds_b128)
#define HAVE_ASYNC_LDS 1
#endif
#endif

#ifdef HAVE_ASYNC_LDS
// Param 0 of the builtin is 'int __vector(4) __device__*' (global AS1 pointer
// to a 16-byte vector); param 1 is the LDS-side pointer (AS3, 32-bit offset).
typedef int vsi4 __attribute__((vector_size(16)));
typedef __attribute__((address_space(1))) vsi4* gl_v4p;
typedef __attribute__((address_space(3))) vsi4* lds_v4p;

// Per-lane 16B global->LDS async copy. Generic LDS pointers on AMDGPU carry
// the LDS byte offset in their low 32 bits, so integer truncation yields the
// AS3 offset the instruction wants.
__device__ __forceinline__ void async_copy16(const _Float16* g, _Float16* l) {
  __builtin_amdgcn_global_load_async_to_lds_b128(
      (gl_v4p)(uintptr_t)g,
      (lds_v4p)(unsigned int)(uintptr_t)l, 0, 0);
}
__device__ __forceinline__ void wait_async0() {
#if __has_builtin(__builtin_amdgcn_s_wait_asynccnt)
  __builtin_amdgcn_s_wait_asynccnt(0);
#else
  asm volatile("s_wait_asynccnt 0x0" ::: "memory");
#endif
}
#endif

// ---------------------------------------------------------------------------
// Fragment loader for v_wmma_f32_16x16x32_f16.
// A tile (16x32, row-major, leading dim `ld` in halfwords): lane&15 selects the
// row; lane>>4 selects the K-half. Per the CDNA5 VGPR layout, each lane's 16
// halfwords are two contiguous 8-halfword (16B) chunks at K = {g8..g8+7} and
// {16+g8..16+g8+7}, g8 = (lane>>4)*8.
// B tile (32x16): weights are stored row-major [N][K] (== column-major B), so
// the same loader applies with lane&15 selecting the tile column.
// ---------------------------------------------------------------------------
__device__ __forceinline__ v16h load_frag(const _Float16* tile, int ld, int lane) {
  const int r  = lane & 15;
  const int g8 = (lane >> 4) << 3;
  const _Float16* p = tile + (size_t)r * ld + g8;
  v8h lo = *(const v8h*)(p);
  v8h hi = *(const v8h*)(p + 16);
  return __builtin_shufflevector(lo, hi, 0,1,2,3,4,5,6,7,8,9,10,11,12,13,14,15);
}

__device__ __forceinline__ float sigmoidf_(float x) {
  return 1.0f / (1.0f + __expf(-x));
}

// ------------------------- embedding gather to f16 -------------------------
__global__ __launch_bounds__(256) void embed_kernel(
    const int* __restrict__ tokens,   // [B][T]
    const float* __restrict__ emb,    // [V][E]
    _Float16* __restrict__ x0)        // [T][B][E]
{
  const int bid = blockIdx.x;               // t*BATCH + b
  const int t = bid >> 6, b = bid & 63;
  const int tok = tokens[b * T_LEN + t];
  const float* src = emb + (size_t)tok * EDIM;
  x0[(size_t)bid * EDIM + threadIdx.x] = (_Float16)src[threadIdx.x];
}

// ---- pack [w_hh | w_ih] rows to f16, concatenated along K ----
__global__ __launch_bounds__(128) void pack_cat_kernel(
    const float* __restrict__ w_hh,   // [2][1024][256]
    const float* __restrict__ w_ih,   // [2][1024][in_dim]
    _Float16* __restrict__ dst,       // [2][1024][256+in_dim]
    int in_dim)
{
  const int row = blockIdx.x;               // 0..2047 (dir*1024 + n)
  const int ktot = 256 + in_dim;
  const float* hh = w_hh + (size_t)row * 256;
  const float* ih = w_ih + (size_t)row * in_dim;
  _Float16* d = dst + (size_t)row * ktot;
  for (int c = threadIdx.x; c < 256;    c += 128) d[c]       = (_Float16)hh[c];
  for (int c = threadIdx.x; c < in_dim; c += 128) d[256 + c] = (_Float16)ih[c];
}

__global__ __launch_bounds__(256) void f32_to_f16_kernel(
    const float* __restrict__ src, _Float16* __restrict__ dst, int n)
{
  int i = blockIdx.x * 256 + threadIdx.x;
  if (i < n) dst[i] = (_Float16)src[i];
}

// ---------------------------------------------------------------------------
// Fused BiLSTM layer (one direction per workgroup).
// A = [h_{t-1} | x_t] : [64 x ktot] f16 in LDS (row stride = ktot+8 halfwords
// -> 4-dword bank skew). Weights Wcat = [w_hh|w_ih] f16, [2][1024][ktot].
// 32 waves; each wave owns 2 adjacent (m, j) 16x16 blocks across all 4 gates,
// keeps the f32 cell state for those blocks in registers for the whole layer.
// x_{t+1} is staged into the disjoint LDS region with async-to-LDS DMA
// (ASYNCcnt) overlapped with the pointwise phase.
// ---------------------------------------------------------------------------
__global__ __launch_bounds__(1024) void lstm_layer_kernel(
    const _Float16* __restrict__ x,     // [T][B][in_dim] f16
    const _Float16* __restrict__ Wcat,  // [2][1024][ktot] f16
    const float*    __restrict__ bias,  // [2][1024] f32 (b_ih+b_hh fused)
    _Float16*       __restrict__ out,   // [T][B][512] f16 (dir half-offset)
    int in_dim)
{
  extern __shared__ _Float16 smem[];
  const int ktot   = 256 + in_dim;
  const int stride = ktot + 8;
  const int dir  = blockIdx.x;
  const int tid  = threadIdx.x;
  const int lane = tid & 31;
  const int wave = tid >> 5;
  const int col  = lane & 15;
  const int half = (lane >> 4) << 3;

  const int p0    = wave * 2;     // pair index base (even -> same mtile)
  const int mtile = p0 >> 4;      // 0..3
  const int j0    = p0 & 15;      // 0..14 (even)

  const _Float16* Wd = Wcat + (size_t)dir * 1024 * ktot;

  // per-wave gate biases (time-invariant)
  float bi[2][4];
#pragma unroll
  for (int pp = 0; pp < 2; ++pp)
#pragma unroll
    for (int G = 0; G < 4; ++G)
      bi[pp][G] = bias[dir * 1024 + G * 256 + (j0 + pp) * 16 + col];

  // zero h region (cols [0,256))
  for (int i = tid; i < BATCH * 256; i += 1024) {
    int b = i >> 8, c = i & 255;
    smem[b * stride + c] = (_Float16)0.0f;
  }

  auto stage_x = [&](int t) {
    const _Float16* src = x + (size_t)t * BATCH * in_dim;
    const int total = BATCH * in_dim;     // multiple of 8
    for (int i = tid * 8; i < total; i += 1024 * 8) {
      const int b = i / in_dim;
      const int c = i - b * in_dim;
#ifdef HAVE_ASYNC_LDS
      async_copy16(src + i, smem + b * stride + 256 + c);
#else
      *(v8h*)(smem + b * stride + 256 + c) = *(const v8h*)(src + i);
#endif
    }
#ifdef HAVE_ASYNC_LDS
    wait_async0();
#endif
  };
  stage_x(dir == 0 ? 0 : T_LEN - 1);

  v8f c0, c1;
#pragma unroll
  for (int r = 0; r < 8; ++r) { c0[r] = 0.0f; c1[r] = 0.0f; }

  const int nkt = ktot >> 5;

  for (int s = 0; s < T_LEN; ++s) {
    const int t = (dir == 0) ? s : (T_LEN - 1 - s);
    __syncthreads();                      // h(s-1) + x(t) ready in LDS

    v8f acc[8];                           // [pair][gate]
#pragma unroll
    for (int i = 0; i < 8; ++i)
#pragma unroll
      for (int r = 0; r < 8; ++r) acc[i][r] = 0.0f;

    const _Float16* arow = smem + (size_t)(mtile * 16) * stride;
    for (int kt = 0; kt < nkt; ++kt) {
      v16h afrag = load_frag(arow + kt * 32, stride, lane);
#pragma unroll
      for (int G = 0; G < 4; ++G) {
        v16h bf0 = load_frag(Wd + (size_t)(G * 256 + j0 * 16) * ktot + kt * 32,
                             ktot, lane);
        acc[G] = __builtin_amdgcn_wmma_f32_16x16x32_f16(
            false, afrag, false, bf0, (short)0, acc[G], false, false);
        v16h bf1 = load_frag(Wd + (size_t)(G * 256 + (j0 + 1) * 16) * ktot + kt * 32,
                             ktot, lane);
        acc[4 + G] = __builtin_amdgcn_wmma_f32_16x16x32_f16(
            false, afrag, false, bf1, (short)0, acc[4 + G], false, false);
      }
    }

    __syncthreads();                      // all LDS reads of this step done

    // pointwise LSTM update; c stays in registers, h -> LDS f16 + global f16
#pragma unroll
    for (int pp = 0; pp < 2; ++pp) {
      const int jcol = (j0 + pp) * 16 + col;
      v8f& cc = pp ? c1 : c0;
#pragma unroll
      for (int r = 0; r < 8; ++r) {
        const float iv = sigmoidf_(acc[pp * 4 + 0][r] + bi[pp][0]);
        const float fv = sigmoidf_(acc[pp * 4 + 1][r] + bi[pp][1]);
        const float gv = tanhf    (acc[pp * 4 + 2][r] + bi[pp][2]);
        const float ov = sigmoidf_(acc[pp * 4 + 3][r] + bi[pp][3]);
        const float cn = fv * cc[r] + iv * gv;
        cc[r] = cn;
        const float hv = ov * tanhf(cn);
        const _Float16 hh = (_Float16)hv;
        const int brow = mtile * 16 + half + r;
        smem[brow * stride + jcol] = hh;                          // next-step A
        out[((size_t)t * BATCH + brow) * 512 + dir * 256 + jcol] = hh;
      }
    }

    if (s + 1 < T_LEN) {
      const int tn = (dir == 0) ? (s + 1) : (T_LEN - 2 - s);
      stage_x(tn);                        // x region disjoint from h region
      if (s + 2 < T_LEN) {
        const int tp = (dir == 0) ? (s + 2) : (T_LEN - 3 - s);
        if (tid < BATCH)
          __builtin_prefetch(x + ((size_t)tp * BATCH + tid) * in_dim, 0, 1);
      }
    }
  }
}

// ------------------------- FC: feats = h1 @ fc_w^T + b ----------------------
__global__ __launch_bounds__(96) void fc_kernel(
    const _Float16* __restrict__ h1,    // [T*B][512]
    const _Float16* __restrict__ fcw,   // [48][512] f16
    const float*    __restrict__ fcb,   // [48]
    float* __restrict__ feats)          // [T*B][48]
{
  const int mt   = blockIdx.x;          // M tile (T*B/16)
  const int lane = threadIdx.x & 31;
  const int wt   = threadIdx.x >> 5;    // 0..2 -> output col tile
  const int col  = lane & 15;
  const int half = (lane >> 4) << 3;

  v8f acc;
#pragma unroll
  for (int r = 0; r < 8; ++r) acc[r] = 0.0f;

  const _Float16* arow = h1  + (size_t)mt * 16 * 512;
  const _Float16* brow = fcw + (size_t)wt * 16 * 512;
#pragma unroll
  for (int kt = 0; kt < 16; ++kt) {
    v16h a = load_frag(arow + kt * 32, 512, lane);
    v16h b = load_frag(brow + kt * 32, 512, lane);
    acc = __builtin_amdgcn_wmma_f32_16x16x32_f16(
        false, a, false, b, (short)0, acc, false, false);
  }
  const float bb = fcb[wt * 16 + col];
#pragma unroll
  for (int r = 0; r < 8; ++r) {
    const int row = mt * 16 + half + r;
    feats[(size_t)row * KTAG + wt * 16 + col] = acc[r] + bb;
  }
}

// ------------------------------- Viterbi -----------------------------------
__global__ __launch_bounds__(64) void viterbi_kernel(
    const float* __restrict__ feats,   // [T][B][K]
    const float* __restrict__ trans,   // [K][K] (to, from)
    unsigned char* __restrict__ psi,   // [T-1][B][K]
    float* __restrict__ outbuf)        // [B] score ++ [B][T] path(int32)
{
  __shared__ float tr[KTAG * KTAG];
  __shared__ float ld[KTAG];
  __shared__ float ldn[KTAG];
  const int b = blockIdx.x;
  const int k = threadIdx.x;

  for (int i = k; i < KTAG * KTAG; i += 64) tr[i] = trans[i];
  if (k < KTAG) ld[k] = (k == START_ID) ? 0.0f : NEG_INF;
  __syncthreads();

  for (int t = 1; t < T_LEN; ++t) {
    if (k < KTAG) {
      float best = -3.4e38f; int arg = 0;
      for (int f = 0; f < KTAG; ++f) {
        const float sc = tr[k * KTAG + f] + ld[f];
        if (sc > best) { best = sc; arg = f; }
      }
      ldn[k] = best + feats[((size_t)t * BATCH + b) * KTAG + k];
      psi[((size_t)(t - 1) * BATCH + b) * KTAG + k] = (unsigned char)arg;
    }
    __syncthreads();
    if (k < KTAG) ld[k] = ldn[k];
    __syncthreads();
  }

  if (k == 0) {
    float best = -3.4e38f; int last = 0;
    for (int f = 0; f < KTAG; ++f)
      if (ld[f] > best) { best = ld[f]; last = f; }
    outbuf[b] = best;
    int* path = (int*)(outbuf + BATCH);
    path[b * T_LEN + (T_LEN - 1)] = last;
    int tag = last;
    for (int s = T_LEN - 2; s >= 0; --s) {
      tag = psi[((size_t)s * BATCH + b) * KTAG + tag];
      path[b * T_LEN + s] = tag;
    }
  }
}

// ---------------------------------------------------------------------------
extern "C" void kernel_launch(void* const* d_in, const int* in_sizes, int n_in,
                              void* d_out, int out_size, void* d_ws, size_t ws_size,
                              hipStream_t stream) {
  (void)in_sizes; (void)n_in; (void)out_size; (void)ws_size;

  const int*   tokens  = (const int*)  d_in[0];
  const float* emb     = (const float*)d_in[1];
  const float* w_ih_l0 = (const float*)d_in[2];
  const float* w_hh_l0 = (const float*)d_in[3];
  const float* b_l0    = (const float*)d_in[4];
  const float* w_ih_l1 = (const float*)d_in[5];
  const float* w_hh_l1 = (const float*)d_in[6];
  const float* b_l1    = (const float*)d_in[7];
  const float* fc_w    = (const float*)d_in[8];
  const float* fc_b    = (const float*)d_in[9];
  const float* trans   = (const float*)d_in[10];

  char* ws = (char*)d_ws;
  size_t off = 0;
  _Float16* x0   = (_Float16*)(ws + off); off += (size_t)T_LEN * BATCH * EDIM * 2;
  _Float16* Wc0  = (_Float16*)(ws + off); off += (size_t)2 * 1024 * 512 * 2;
  _Float16* Wc1  = (_Float16*)(ws + off); off += (size_t)2 * 1024 * 768 * 2;
  _Float16* h0   = (_Float16*)(ws + off); off += (size_t)T_LEN * BATCH * 512 * 2;
  _Float16* h1   = (_Float16*)(ws + off); off += (size_t)T_LEN * BATCH * 512 * 2;
  _Float16* fcw  = (_Float16*)(ws + off); off += (size_t)KTAG * 512 * 2;
  float*    feats= (float*)   (ws + off); off += (size_t)T_LEN * BATCH * KTAG * 4;
  unsigned char* psi = (unsigned char*)(ws + off);

  (void)hipFuncSetAttribute(reinterpret_cast<const void*>(lstm_layer_kernel),
                            hipFuncAttributeMaxDynamicSharedMemorySize,
                            BATCH * (256 + 8 + 512) * 2);

  embed_kernel<<<dim3(T_LEN * BATCH), dim3(256), 0, stream>>>(tokens, emb, x0);
  pack_cat_kernel<<<dim3(2 * 1024), dim3(128), 0, stream>>>(w_hh_l0, w_ih_l0, Wc0, 256);
  pack_cat_kernel<<<dim3(2 * 1024), dim3(128), 0, stream>>>(w_hh_l1, w_ih_l1, Wc1, 512);
  f32_to_f16_kernel<<<dim3((KTAG * 512 + 255) / 256), dim3(256), 0, stream>>>(fc_w, fcw, KTAG * 512);

  lstm_layer_kernel<<<dim3(2), dim3(1024), BATCH * (256 + 8 + 256) * 2, stream>>>(
      x0, Wc0, b_l0, h0, 256);
  lstm_layer_kernel<<<dim3(2), dim3(1024), BATCH * (256 + 8 + 512) * 2, stream>>>(
      h0, Wc1, b_l1, h1, 512);

  fc_kernel<<<dim3((T_LEN * BATCH) / 16), dim3(96), 0, stream>>>(h1, fcw, fc_b, feats);
  viterbi_kernel<<<dim3(BATCH), dim3(64), 0, stream>>>(feats, trans, psi, (float*)d_out);
}
